// OptionDuelingDQN_11020886082223
// MI455X (gfx1250) — compile-verified, matrix operand order checked by reading, not run on hardware
//
#include <hip/hip_runtime.h>
#include <hip/hip_fp16.h>

// B=4096, C=4, HW=64, A=32, E=4, H1=1024, HID=512
// conv1: 4->32 k8 s4 -> 15x15 (pad 240 pos, NHWC)
// conv2: 32->64 k4 s2 -> 6x6  (pad 48 pos, NHWC)
// conv3: 64->64 k3 s1 -> 4x4  -> h3 [B][1024] (feature = c*16 + y*4 + x)

typedef __attribute__((ext_vector_type(16))) _Float16 v16h;
typedef __attribute__((ext_vector_type(8)))  _Float16 v8h;
typedef __attribute__((ext_vector_type(8)))  float    v8f;
typedef __attribute__((ext_vector_type(4)))  unsigned int u32x4;
typedef __attribute__((ext_vector_type(8)))  int      i32x8;
typedef __attribute__((ext_vector_type(4)))  int      i32x4;

#if defined(__gfx1250__) && __has_builtin(__builtin_amdgcn_tensor_load_to_lds) && \
    __has_builtin(__builtin_amdgcn_s_wait_tensorcnt)
#define USE_TDM 1
#else
#define USE_TDM 0
#endif

// lane L<16 holds K offsets {0..7,16..23}; lane L>=16 holds {8..15,24..31}
__device__ __forceinline__ int frag_k(int lane, int e) {
  return ((e < 8) ? e : (e + 8)) + ((lane & 16) ? 8 : 0);
}

__device__ __forceinline__ v8f wmma_f16(v16h a, v16h b, v8f c) {
  return __builtin_amdgcn_wmma_f32_16x16x32_f16(false, a, false, b, (short)0, c,
                                                false, false);
}

__device__ __forceinline__ v16h make_a(v8h lo, v8h hi) {
  v16h a;
  #pragma unroll
  for (int i = 0; i < 8; ++i) { a[i] = lo[i]; a[8 + i] = hi[i]; }
  return a;
}

// ---- Weight repack into fragment-major f16 tiles ----
// dst: [exp][ktile*Ntiles + ntile][lane(32)][elem(16)]
// mode 0: src[k*Nsrc + n]                      (MLP, K-major, N contiguous)
// mode 1: src[n*K + k]                         (conv1: k = c*64 + kh*8 + kw)
// mode 2: conv2: k = s*32 + c -> src[n*512 + c*16 + s]   (s = kh*4+kw)
// mode 3: conv3: k = s*64 + c -> src[n*576 + c*9 + s]    (s = kh*3+kw)
__global__ void pack_b_kernel(const float* __restrict__ src, _Float16* __restrict__ dst,
                              int K, int N, int Nsrc, int mode, int nExp) {
  int tid = blockIdx.x * blockDim.x + threadIdx.x;
  int Ntiles = N >> 4;
  int perExp = (K >> 5) * Ntiles * 512;
  if (tid >= perExp * nExp) return;
  int ei = tid / perExp;
  int r0 = tid - ei * perExp;
  int tile = r0 >> 9;
  int r = r0 & 511;
  int lane = r >> 4, e = r & 15;
  int kt = tile / Ntiles, nt = tile - kt * Ntiles;
  int k = kt * 32 + frag_k(lane, e);
  int n = nt * 16 + (lane & 15);
  const float* s = src + (size_t)ei * K * Nsrc;
  float v = 0.f;
  if (n < Nsrc) {
    if (mode == 0) v = s[(size_t)k * Nsrc + n];
    else if (mode == 1) v = s[(size_t)n * K + k];
    else if (mode == 2) { int sp = k >> 5, c = k & 31; v = s[(size_t)n * 512 + c * 16 + sp]; }
    else                { int sp = k >> 6, c = k & 63; v = s[(size_t)n * 576 + c * 9 + sp]; }
  }
  dst[tid] = (_Float16)v;
}

// ---- Routing ----
// meta: [0..3]=counts [4..7]=cursor [8..11]=base [12]=ntiles [13]=always 0 (opaque)
__global__ void init_route_kernel(int* __restrict__ perm, int* __restrict__ meta) {
  int t = blockIdx.x * blockDim.x + threadIdx.x;
  if (t < 4160) perm[t] = 0;
  if (t < 16) meta[t] = 0;
}

__global__ void route_kernel(const float* __restrict__ x, const float* __restrict__ bnd,
                             int* __restrict__ idx, int* __restrict__ meta) {
  int b = blockIdx.x * blockDim.x + threadIdx.x;
  if (b >= 4096) return;
  float r = x[(size_t)b * 16384 + 3 * 4096];   // x[b, C-1, 0, 0]
  int e = (r > bnd[0]) + (r > bnd[1]) + (r > bnd[2]);
  idx[b] = e;
  atomicAdd(&meta[e], 1);
}

__global__ void scan_kernel(int* __restrict__ meta, int* __restrict__ te,
                            int* __restrict__ tms, int* __restrict__ tme) {
  if (threadIdx.x != 0 || blockIdx.x != 0) return;
  int mstart = 0, tt = 0;
  for (int e = 0; e < 4; ++e) {
    meta[8 + e] = mstart;
    int cnt = meta[e];
    int ntl = (cnt + 15) >> 4;
    for (int i = 0; i < ntl; ++i) {
      te[tt] = e; tms[tt] = mstart + i * 16; tme[tt] = mstart + cnt; ++tt;
    }
    mstart += ntl * 16;
  }
  meta[12] = tt;
}

__global__ void scatter_kernel(const int* __restrict__ idx, int* __restrict__ meta,
                               int* __restrict__ perm) {
  int b = blockIdx.x * blockDim.x + threadIdx.x;
  if (b >= 4096) return;
  int e = idx[b];
  int pos = meta[8 + e] + atomicAdd(&meta[4 + e], 1);
  perm[pos] = b;
}

// ---- conv1: M=B*225(pad240), N=32 (both tiles per wave), K=256 ----
__global__ __launch_bounds__(128)
void conv1_kernel(const float* __restrict__ x, const float* __restrict__ cb1,
                  const _Float16* __restrict__ Wf, _Float16* __restrict__ h1) {
  const int wave = threadIdx.x >> 5, lane = threadIdx.x & 31;
  const int mtile = blockIdx.x * 4 + wave;          // 15 tiles per image
  const int img = mtile / 15, ptile = mtile - img * 15;
  const int rr = lane & 15;
  int p = ptile * 16 + rr; if (p > 224) p = 224;
  const int py = p / 15, px = p - py * 15;
  const int koff = (lane & 16) ? 8 : 0;
  v8f acc0 = {}, acc1 = {};
  for (int kt = 0; kt < 8; ++kt) {
    v16h a;
    #pragma unroll
    for (int run = 0; run < 2; ++run) {
      int ks = kt * 32 + koff + run * 16;           // 8-aligned: one (c,kh), kw=0..7
      int c = ks >> 6, kh = (ks >> 3) & 7;
      const float* src = x + (((size_t)img * 4 + c) * 64 + (py * 4 + kh)) * 64 + px * 4;
      float4 f0 = *(const float4*)(src);
      float4 f1 = *(const float4*)(src + 4);
      int e0 = run * 8;
      a[e0+0]=(_Float16)f0.x; a[e0+1]=(_Float16)f0.y; a[e0+2]=(_Float16)f0.z; a[e0+3]=(_Float16)f0.w;
      a[e0+4]=(_Float16)f1.x; a[e0+5]=(_Float16)f1.y; a[e0+6]=(_Float16)f1.z; a[e0+7]=(_Float16)f1.w;
    }
    v16h b0 = *(const v16h*)(Wf + ((size_t)(kt * 2 + 0) * 32 + lane) * 16);
    v16h b1 = *(const v16h*)(Wf + ((size_t)(kt * 2 + 1) * 32 + lane) * 16);
    acc0 = wmma_f16(a, b0, acc0);
    acc1 = wmma_f16(a, b1, acc1);
  }
  const float bias0 = cb1[rr], bias1 = cb1[16 + rr];
  #pragma unroll
  for (int i = 0; i < 8; ++i) {
    int pp = ptile * 16 + i + ((lane >> 4) << 3);   // < 240 always (padded)
    _Float16* dst = h1 + ((size_t)img * 240 + pp) * 32;
    float v0 = acc0[i] + bias0; v0 = v0 > 0.f ? v0 : 0.f;
    float v1 = acc1[i] + bias1; v1 = v1 > 0.f ? v1 : 0.f;
    dst[rr] = (_Float16)v0;
    dst[16 + rr] = (_Float16)v1;
  }
}

// ---- conv2: M=B*36(pad48), N=64 (2 tiles/wave), K=512 (k = (kh*4+kw)*32 + c) ----
__global__ __launch_bounds__(128)
void conv2_kernel(const _Float16* __restrict__ h1, const float* __restrict__ cb2,
                  const _Float16* __restrict__ Wf, _Float16* __restrict__ h2) {
  const int wave = threadIdx.x >> 5, lane = threadIdx.x & 31;
  const int mtile = blockIdx.x * 4 + wave;          // 3 tiles per image
  const int img = mtile / 3, ptile = mtile - img * 3;
  const int ng = blockIdx.y;                        // 0..1 -> ntiles {2ng, 2ng+1}
  const int rr = lane & 15;
  int p = ptile * 16 + rr; if (p > 35) p = 35;
  const int py = p / 6, px = p - py * 6;
  const int koff = (lane & 16) ? 8 : 0;
  const _Float16* src = h1 + (size_t)img * 240 * 32;
  v8f acc0 = {}, acc1 = {};
  for (int kt = 0; kt < 16; ++kt) {                 // s = kt = kh*4+kw
    int kh = kt >> 2, kw = kt & 3;
    const _Float16* row = src + ((size_t)((py * 2 + kh) * 15 + (px * 2 + kw))) * 32;
    v8h t0 = *(const v8h*)(row + koff);
    v8h t1 = *(const v8h*)(row + koff + 16);
    v16h a = make_a(t0, t1);
    v16h b0 = *(const v16h*)(Wf + ((size_t)(kt * 4 + ng * 2 + 0) * 32 + lane) * 16);
    v16h b1 = *(const v16h*)(Wf + ((size_t)(kt * 4 + ng * 2 + 1) * 32 + lane) * 16);
    acc0 = wmma_f16(a, b0, acc0);
    acc1 = wmma_f16(a, b1, acc1);
  }
  const float bias0 = cb2[ng * 32 + rr], bias1 = cb2[ng * 32 + 16 + rr];
  #pragma unroll
  for (int i = 0; i < 8; ++i) {
    int pp = ptile * 16 + i + ((lane >> 4) << 3);   // < 48 always (padded)
    _Float16* dst = h2 + ((size_t)img * 48 + pp) * 64 + ng * 32;
    float v0 = acc0[i] + bias0; v0 = v0 > 0.f ? v0 : 0.f;
    float v1 = acc1[i] + bias1; v1 = v1 > 0.f ? v1 : 0.f;
    dst[rr] = (_Float16)v0;
    dst[16 + rr] = (_Float16)v1;
  }
}

// ---- conv3: M=B*16, N=64 (2 tiles/wave), K=576 (k = (kh*3+kw)*64 + c) ----
__global__ __launch_bounds__(128)
void conv3_kernel(const _Float16* __restrict__ h2, const float* __restrict__ cb3,
                  const _Float16* __restrict__ Wf, _Float16* __restrict__ h3) {
  const int wave = threadIdx.x >> 5, lane = threadIdx.x & 31;
  const int img = blockIdx.x * 4 + wave;
  const int ng = blockIdx.y;                        // 0..1
  const int rr = lane & 15;
  const int py = rr >> 2, px = rr & 3;
  const int koff = (lane & 16) ? 8 : 0;
  const _Float16* src = h2 + (size_t)img * 48 * 64;
  v8f acc0 = {}, acc1 = {};
  for (int kt = 0; kt < 18; ++kt) {                 // 32 k per step within s = kt>>1
    int s = kt >> 1;
    int kh = s / 3, kw = s - kh * 3;
    const _Float16* row = src + ((size_t)((py + kh) * 6 + (px + kw))) * 64 + (kt & 1) * 32;
    v8h t0 = *(const v8h*)(row + koff);
    v8h t1 = *(const v8h*)(row + koff + 16);
    v16h a = make_a(t0, t1);
    v16h b0 = *(const v16h*)(Wf + ((size_t)(kt * 4 + ng * 2 + 0) * 32 + lane) * 16);
    v16h b1 = *(const v16h*)(Wf + ((size_t)(kt * 4 + ng * 2 + 1) * 32 + lane) * 16);
    acc0 = wmma_f16(a, b0, acc0);
    acc1 = wmma_f16(a, b1, acc1);
  }
  const float bias0 = cb3[ng * 32 + rr], bias1 = cb3[ng * 32 + 16 + rr];
  const int hi8 = (lane >> 4) << 3;
  v8h o0, o1;
  #pragma unroll
  for (int i = 0; i < 8; ++i) {
    float v0 = acc0[i] + bias0; v0 = v0 > 0.f ? v0 : 0.f;
    float v1 = acc1[i] + bias1; v1 = v1 > 0.f ? v1 : 0.f;
    o0[i] = (_Float16)v0;
    o1[i] = (_Float16)v1;
  }
  // feature = ch*16 + p ; 8 consecutive positions -> one b128 store each
  *(v8h*)(h3 + (size_t)img * 1024 + (ng * 32 + rr) * 16 + hi8) = o0;
  *(v8h*)(h3 + (size_t)img * 1024 + (ng * 32 + 16 + rr) * 16 + hi8) = o1;
}

// ---- mlp1: routed, both nets. 8 waves share one TDM-gathered A tile in LDS ----
// grid: (260, 4): net = y>>1 ; slot = (y&1)*8 + wave ; nt0 = slot*2
__global__ __launch_bounds__(256)
void mlp1_kernel(const _Float16* __restrict__ h3,
                 const _Float16* __restrict__ aW1f, const float* __restrict__ ab1,
                 const _Float16* __restrict__ vW1f, const float* __restrict__ vb1,
                 const int* __restrict__ perm, const int* __restrict__ meta,
                 const int* __restrict__ te, const int* __restrict__ tms,
                 _Float16* __restrict__ za, _Float16* __restrict__ zv) {
  __shared__ _Float16 smem[16 * 1024];            // 32 KB gathered A tile
  const int t = blockIdx.x;
  if (t >= meta[12]) return;
  const int tid = threadIdx.x;
  const int wave = tid >> 5, lane = tid & 31;
  const int e = te[t], mstart = tms[t];

#if USE_TDM
  if (wave == 0) {
    const int* pr = perm + mstart;
    // D# group2/3: 16 packed 16-bit row indices (gather mode)
    i32x4 g2, g3;
    g2[0] = (pr[0] & 0xFFFF) | (pr[1] << 16);
    g2[1] = (pr[2] & 0xFFFF) | (pr[3] << 16);
    g2[2] = (pr[4] & 0xFFFF) | (pr[5] << 16);
    g2[3] = (pr[6] & 0xFFFF) | (pr[7] << 16);
    g3[0] = (pr[8] & 0xFFFF) | (pr[9] << 16);
    g3[1] = (pr[10] & 0xFFFF) | (pr[11] << 16);
    g3[2] = (pr[12] & 0xFFFF) | (pr[13] << 16);
    g3[3] = (pr[14] & 0xFFFF) | (pr[15] << 16);
    // D# group0: count=1, gather_mode=1 (16-bit idx), lds_addr=0 (smem), global=h3, type=2
    unsigned long long ga = (unsigned long long)(const void*)h3;
    u32x4 g0;
    g0[0] = 0x80000001u;
    g0[1] = 0u;
    g0[2] = (unsigned)(ga & 0xFFFFFFFFu);
    g0[3] = (unsigned)((ga >> 32) & 0x01FFFFFFu) | (2u << 30);
    // D# group1: data_size=2B; tensor_dim0=1024; tensor_dim1=4096; tile_dim0=1024;
    //            tile_dim1=16 (valid indices); tensor_dim0_stride=1024
    i32x8 g1;
    g1[0] = 0x00010000;
    g1[1] = (int)(1024u << 16);
    g1[2] = (int)(4096u << 16);
    g1[3] = (int)(1024u << 16);
    g1[4] = 16;
    g1[5] = 1024;
    g1[6] = 0;
    g1[7] = 0;
    i32x8 g4 = {0, 0, 0, 0, 0, 0, 0, 0};       // 6-arg toolchain: extra zero group
    __builtin_amdgcn_tensor_load_to_lds(g0, g1, g2, g3, g4, 0);
    __builtin_amdgcn_s_wait_tensorcnt(0);
    asm volatile("" ::: "memory");             // TDM wrote LDS: block undef-folding
  }
  // opaque never-taken store: keeps smem 'written' for alias analysis
  // (meta[13] is zeroed at runtime but unknowable at compile time)
  if (meta[13] != 0) smem[lane] = (_Float16)1.0f;
#else
  // cooperative fallback: 16 rows x 1024 halves, 16B chunks
  for (int i = tid; i < 16 * 128; i += 256) {
    int row = i >> 7, ko = (i & 127) * 8;
    *(v8h*)(smem + row * 1024 + ko) =
        *(const v8h*)(h3 + (size_t)perm[mstart + row] * 1024 + ko);
  }
#endif
  __syncthreads();

  const int net = blockIdx.y >> 1;                 // 0=a, 1=v
  const int nt0 = (((blockIdx.y & 1) << 3) + wave) * 2;  // 0,2,..,30
  const int rr = lane & 15;
  const _Float16* Wf = (net ? vW1f : aW1f) + (size_t)e * 32 * 32 * 512;
  const float* bias = (net ? vb1 : ab1) + e * 512;
  _Float16* z = net ? zv : za;
  const int koff = (lane & 16) ? 8 : 0;
  const _Float16* arow = smem + rr * 1024;
  v8f acc0 = {}, acc1 = {};
  for (int kt = 0; kt < 32; ++kt) {
    v8h t0 = *(const v8h*)(arow + kt * 32 + koff);          // ds_load_b128
    v8h t1 = *(const v8h*)(arow + kt * 32 + koff + 16);
    v16h a = make_a(t0, t1);
    v16h b0 = *(const v16h*)(Wf + ((size_t)(kt * 32 + nt0 + 0) * 32 + lane) * 16);
    v16h b1 = *(const v16h*)(Wf + ((size_t)(kt * 32 + nt0 + 1) * 32 + lane) * 16);
    acc0 = wmma_f16(a, b0, acc0);
    acc1 = wmma_f16(a, b1, acc1);
  }
  const float bv0 = bias[nt0 * 16 + rr], bv1 = bias[(nt0 + 1) * 16 + rr];
  #pragma unroll
  for (int i = 0; i < 8; ++i) {
    int gm = mstart + i + ((lane >> 4) << 3);
    float v0 = acc0[i] + bv0; v0 = v0 > 0.f ? v0 : 0.f;
    float v1 = acc1[i] + bv1; v1 = v1 > 0.f ? v1 : 0.f;
    z[(size_t)gm * 512 + nt0 * 16 + rr] = (_Float16)v0;
    z[(size_t)gm * 512 + (nt0 + 1) * 16 + rr] = (_Float16)v1;
  }
}

// ---- mlp2: fused heads + dueling combine, scatter to d_out ----
__global__ __launch_bounds__(32)
void mlp2_kernel(const _Float16* __restrict__ za, const _Float16* __restrict__ zv,
                 const _Float16* __restrict__ aW2f, const float* __restrict__ ab2,
                 const _Float16* __restrict__ vW2f, const float* __restrict__ vb2,
                 const int* __restrict__ perm, const int* __restrict__ meta,
                 const int* __restrict__ te, const int* __restrict__ tms,
                 const int* __restrict__ tme, float* __restrict__ out) {
  const int t = blockIdx.x;
  if (t >= meta[12]) return;
  const int lane = threadIdx.x & 31;
  const int e = te[t], mstart = tms[t], mend = tme[t];
  const int rr = lane & 15;
  const int koff = (lane & 16) ? 8 : 0;
  const _Float16* zarow = za + (size_t)(mstart + rr) * 512;
  const _Float16* zvrow = zv + (size_t)(mstart + rr) * 512;
  const _Float16* Wa = aW2f + (size_t)e * 32 * 512;   // 32 tiles (16kt x 2nt)
  const _Float16* Wv = vW2f + (size_t)e * 16 * 512;   // 16 tiles (16kt x 1nt)
  v8f ca0 = {}, ca1 = {}, cv = {};
  for (int kt = 0; kt < 16; ++kt) {
    v8h t0 = *(const v8h*)(zarow + kt * 32 + koff);
    v8h t1 = *(const v8h*)(zarow + kt * 32 + koff + 16);
    v16h a = make_a(t0, t1);
    v16h b0 = *(const v16h*)(Wa + ((size_t)(kt * 2 + 0) * 32 + lane) * 16);
    v16h b1 = *(const v16h*)(Wa + ((size_t)(kt * 2 + 1) * 32 + lane) * 16);
    ca0 = wmma_f16(a, b0, ca0);
    ca1 = wmma_f16(a, b1, ca1);
    v8h s0 = *(const v8h*)(zvrow + kt * 32 + koff);
    v8h s1 = *(const v8h*)(zvrow + kt * 32 + koff + 16);
    v16h av = make_a(s0, s1);
    v16h bv = *(const v16h*)(Wv + ((size_t)kt * 32 + lane) * 16);
    cv = wmma_f16(av, bv, cv);
  }
  const float biasA0 = ab2[e * 32 + rr];
  const float biasA1 = ab2[e * 32 + 16 + rr];
  const float biasV = vb2[e];
  #pragma unroll
  for (int i = 0; i < 8; ++i) {
    float ya0 = ca0[i] + biasA0;
    float ya1 = ca1[i] + biasA1;
    float s = ya0 + ya1;                 // per-row sum over 32 cols within 16-lane half
    s += __shfl_xor(s, 1, 16);
    s += __shfl_xor(s, 2, 16);
    s += __shfl_xor(s, 4, 16);
    s += __shfl_xor(s, 8, 16);
    float mean = s * (1.f / 32.f);
    float ys = __shfl(cv[i], lane & 16, 32) + biasV;   // n==0 column of own half
    int gm = mstart + i + ((lane >> 4) << 3);
    if (gm < mend) {
      int b = perm[gm];
      out[(size_t)b * 32 + rr] = ya0 - mean + ys;
      out[(size_t)b * 32 + 16 + rr] = ya1 - mean + ys;
    }
  }
}

extern "C" void kernel_launch(void* const* d_in, const int* in_sizes, int n_in,
                              void* d_out, int out_size, void* d_ws, size_t ws_size,
                              hipStream_t stream) {
  (void)in_sizes; (void)n_in; (void)out_size; (void)ws_size;
  const float* x   = (const float*)d_in[0];
  const float* cW1 = (const float*)d_in[1];
  const float* cb1 = (const float*)d_in[2];
  const float* cW2 = (const float*)d_in[3];
  const float* cb2 = (const float*)d_in[4];
  const float* cW3 = (const float*)d_in[5];
  const float* cb3 = (const float*)d_in[6];
  const float* aW1 = (const float*)d_in[7];
  const float* ab1 = (const float*)d_in[8];
  const float* aW2 = (const float*)d_in[9];
  const float* ab2 = (const float*)d_in[10];
  const float* vW1 = (const float*)d_in[11];
  const float* vb1 = (const float*)d_in[12];
  const float* vW2 = (const float*)d_in[13];
  const float* vb2 = (const float*)d_in[14];
  const float* bnd = (const float*)d_in[15];
  float* out = (float*)d_out;

  char* base = (char*)d_ws;
  size_t off = 0;
  auto alloc = [&](size_t bytes) -> void* {
    void* p = base + off;
    off = (off + bytes + 255) & ~(size_t)255;
    return p;
  };

  _Float16* cW1f = (_Float16*)alloc((size_t)16 * 512 * 2);
  _Float16* cW2f = (_Float16*)alloc((size_t)64 * 512 * 2);
  _Float16* cW3f = (_Float16*)alloc((size_t)72 * 512 * 2);
  _Float16* aW1f = (_Float16*)alloc((size_t)4 * 1024 * 512 * 2);
  _Float16* vW1f = (_Float16*)alloc((size_t)4 * 1024 * 512 * 2);
  _Float16* aW2f = (_Float16*)alloc((size_t)4 * 512 * 32 * 2);
  _Float16* vW2f = (_Float16*)alloc((size_t)4 * 512 * 16 * 2);
  _Float16* h1   = (_Float16*)alloc((size_t)4096 * 240 * 32 * 2);  // NHWC padded
  _Float16* h2   = (_Float16*)alloc((size_t)4096 * 48 * 64 * 2);   // NHWC padded
  _Float16* h3   = (_Float16*)alloc((size_t)4096 * 1024 * 2);
  _Float16* za   = (_Float16*)alloc((size_t)4160 * 512 * 2);
  _Float16* zv   = (_Float16*)alloc((size_t)4160 * 512 * 2);
  int* idxArr = (int*)alloc(4096 * 4);
  int* perm   = (int*)alloc(4160 * 4);
  int* meta   = (int*)alloc(64);
  int* te     = (int*)alloc(260 * 4);
  int* tms    = (int*)alloc(260 * 4);
  int* tme    = (int*)alloc(260 * 4);

  auto nb = [](int n, int bs) { return (n + bs - 1) / bs; };

  // Weight repack (fragment-major f16, per-layer K remap)
  pack_b_kernel<<<nb(8192, 256), 256, 0, stream>>>(cW1, cW1f, 256, 32, 32, 1, 1);
  pack_b_kernel<<<nb(32768, 256), 256, 0, stream>>>(cW2, cW2f, 512, 64, 64, 2, 1);
  pack_b_kernel<<<nb(36864, 256), 256, 0, stream>>>(cW3, cW3f, 576, 64, 64, 3, 1);
  pack_b_kernel<<<nb(2097152, 256), 256, 0, stream>>>(aW1, aW1f, 1024, 512, 512, 0, 4);
  pack_b_kernel<<<nb(2097152, 256), 256, 0, stream>>>(vW1, vW1f, 1024, 512, 512, 0, 4);
  pack_b_kernel<<<nb(65536, 256), 256, 0, stream>>>(aW2, aW2f, 512, 32, 32, 0, 4);
  pack_b_kernel<<<nb(32768, 256), 256, 0, stream>>>(vW2, vW2f, 512, 16, 1, 0, 4);

  // Routing (bucket batches by expert into expert-uniform 16-row tiles)
  init_route_kernel<<<nb(4160, 256), 256, 0, stream>>>(perm, meta);
  route_kernel<<<nb(4096, 256), 256, 0, stream>>>(x, bnd, idxArr, meta);
  scan_kernel<<<1, 1, 0, stream>>>(meta, te, tms, tme);
  scatter_kernel<<<nb(4096, 256), 256, 0, stream>>>(idxArr, meta, perm);

  // Conv trunk (implicit GEMM via WMMA, f16 NHWC activations)
  conv1_kernel<<<dim3(15360), 128, 0, stream>>>(x, cb1, cW1f, h1);
  conv2_kernel<<<dim3(3072, 2), 128, 0, stream>>>(h1, cb2, cW2f, h2);
  conv3_kernel<<<dim3(1024, 2), 128, 0, stream>>>(h2, cb3, cW3f, h3);

  // Routed MLPs (TDM row-gather into LDS) + dueling combine
  mlp1_kernel<<<dim3(260, 4), 256, 0, stream>>>(h3, aW1f, ab1, vW1f, vb1,
                                                perm, meta, te, tms, za, zv);
  mlp2_kernel<<<dim3(260, 1), 32, 0, stream>>>(za, zv, aW2f, ab2, vW2f, vb2,
                                               perm, meta, te, tms, tme, out);
}
